// Attention_27358941675773
// MI455X (gfx1250) — matrix operand order repaired and checked by reading, not compiled, not run on hardware
//
#include <hip/hip_runtime.h>

typedef __attribute__((ext_vector_type(16))) __bf16 v16bf;
typedef __attribute__((ext_vector_type(8)))  float  v8f;

#define BATCH 8
#define NSEQ  4096
#define LSEQ  256
#define HDIM  128
#define WCOLS 512              // fused weight: [Q(128) | K(128) | V(256)]
#define INV_SCALER 0.00390625f // 1/256, folded into Q weights (exact: power of two)
#define KT    64               // keys per flash iteration
#define KTLD  (HDIM + 8)       // kt row stride (elems)
#define PTLD  (KT + 8)         // pt/vtT row stride (elems)

// ---------------- WMMA fragment helpers (CDNA5 wave32, ISA 7.12.2) ----------------
// A 16x32 bf16 (this code's k order): lane<16 -> row=lane, K {0..15}; lane>=16 -> row=lane-16, K {16..31}
__device__ inline v16bf load_a_frag(const __bf16* p, int ld, int lane) {
  int row = lane & 15;
  int hi  = (lane >> 4) * 16;
  const __bf16* s = p + (size_t)row * ld;
  union { uint4 u[2]; v16bf v; } x;
  x.u[0] = *(const uint4*)(s + hi);        // K = hi .. hi+7
  x.u[1] = *(const uint4*)(s + hi + 8);    // K = hi+8 .. hi+15
  return x.v;
}

// B 32x16 where B[k][n] = src[n*ld + k]: N-major, K-contiguous source tile.
// Per lane this is one contiguous 32B run -> 2x ds_load_b128.
__device__ inline v16bf load_b_frag_T(const __bf16* p, int ld, int lane) {
  int col = lane & 15;
  int hi  = (lane >> 4) * 16;
  const uint4* s = (const uint4*)(p + (size_t)col * ld + hi);
  union { uint4 u[2]; v16bf v; } x;
  x.u[0] = s[0];
  x.u[1] = s[1];
  return x.v;
}

__device__ inline v8f wmma_bf16(v16bf a, v16bf b, v8f c) {
  return __builtin_amdgcn_wmma_f32_16x16x32_bf16(false, a, false, b, (short)0, c, false, false);
}

// Async global->LDS 16B copy (GLOBAL_LOAD_ASYNC_TO_LDS_B128, ASYNCcnt-tracked).
// ldsaddr: 32-bit LDS byte address (low 32 bits of a generic __shared__ pointer);
// gaddr:   64-bit global address in a VGPR pair (GV addressing mode).
__device__ inline void async_copy_b128(unsigned ldsaddr, const void* gaddr) {
  asm volatile("global_load_async_to_lds_b128 %0, %1, off"
               :: "v"(ldsaddr), "v"(gaddr) : "memory");
}
__device__ inline void wait_asynccnt0() {
  asm volatile("s_wait_asynccnt 0x0" ::: "memory");
}

// ---------------- Kernel 1: fused weight, stored TRANSPOSED: Wt[c][l], c in [0,512), l in [0,256) ----
// cols of original W: [Q(0..127)*1/256 | K(128..255) | V=Vd@Vu (256..511)]
// The attention score scale 1/256 = 2^-8 is folded into Q here (exact power-of-two scale).
__global__ __launch_bounds__(256) void prep_kernel(const float* __restrict__ Q,
                                                   const float* __restrict__ K,
                                                   const float* __restrict__ Vd,
                                                   const float* __restrict__ Vu,
                                                   __bf16* __restrict__ Wt) {
  int idx = blockIdx.x * 256 + threadIdx.x;
  if (idx >= LSEQ * WCOLS) return;
  int l = idx >> 9;         // row in [0,256)
  int c = idx & (WCOLS - 1);
  float val;
  if (c < HDIM) {
    val = Q[l * HDIM + c] * INV_SCALER;
  } else if (c < 2 * HDIM) {
    val = K[l * HDIM + (c - HDIM)];
  } else {
    int m = c - 2 * HDIM;   // V[l][m] = sum_h Vd[l][h] * Vu[h][m]
    float s = 0.f;
#pragma unroll 8
    for (int h = 0; h < HDIM; ++h) s = fmaf(Vd[l * HDIM + h], Vu[h * LSEQ + m], s);
    val = s;
  }
  Wt[(size_t)c * LSEQ + l] = (__bf16)val;   // transposed store: K-contiguous rows
}

// ---------------- Kernel 2: projections GEMM [B*N x 256] @ [256 x 512] -> q,k,v (bf16) ----------------
__global__ __launch_bounds__(256) void proj_kernel(const float* __restrict__ x,
                                                   const __bf16* __restrict__ Wt,
                                                   __bf16* __restrict__ qb,
                                                   __bf16* __restrict__ kb,
                                                   __bf16* __restrict__ vb) {
  __shared__ __bf16 xs[128][40];    // 128x32 tile of x (bf16), +8 pad -> rows 80B (16B aligned)
  __shared__ __bf16 wtT[64][40];    // 64 output cols x 32 K values, K-contiguous
  int tid = threadIdx.x;
  int lane = tid & 31, wave = tid >> 5;
  int m0 = blockIdx.x * 128;
  int n0 = blockIdx.y * 64;

  v8f zero = {};
  v8f acc[4];
#pragma unroll
  for (int t = 0; t < 4; ++t) acc[t] = zero;

  for (int k0 = 0; k0 < LSEQ; k0 += 32) {
    // stage x tile, converting f32 -> bf16 (16 elems / thread)
    {
      int r = tid >> 1, c0 = (tid & 1) * 16;
      const float* src = x + (size_t)(m0 + r) * LSEQ + k0 + c0;
#pragma unroll
      for (int i = 0; i < 16; ++i) xs[r][c0 + i] = (__bf16)src[i];
    }
    // stage Wt tile: 64 rows x 32 K, pure 16B vector copies (4 threads / row)
    {
      int r = tid >> 2, c0 = (tid & 3) * 8;
      *(uint4*)&wtT[r][c0] = *(const uint4*)(Wt + (size_t)(n0 + r) * LSEQ + k0 + c0);
    }
    __syncthreads();
    v16bf a = load_a_frag(&xs[wave * 16][0], 40, lane);
#pragma unroll
    for (int t = 0; t < 4; ++t) {
      v16bf b = load_b_frag_T(&wtT[t * 16][0], 40, lane);
      acc[t] = wmma_bf16(a, b, acc[t]);
    }
    __syncthreads();
  }

  // Epilogue scatter. The q/k/v split is UNIFORM per block (decided by n0 alone),
  // so branch once at block scope; all stores below are base + immediate offset.
  int col = lane & 15, hi = (lane >> 4) * 8;
  int rowb = m0 + wave * 16 + hi;              // rows rowb .. rowb+7 (global row = b*NSEQ + n)
  if (n0 < HDIM) {                             // pure-Q block
    __bf16* dst = qb + (size_t)rowb * HDIM + n0 + col;
#pragma unroll
    for (int t = 0; t < 4; ++t)
#pragma unroll
      for (int r = 0; r < 8; ++r)
        dst[(size_t)r * HDIM + t * 16] = (__bf16)acc[t][r];
  } else if (n0 < 2 * HDIM) {                  // pure-K block
    __bf16* dst = kb + (size_t)rowb * HDIM + (n0 - HDIM) + col;
#pragma unroll
    for (int t = 0; t < 4; ++t)
#pragma unroll
      for (int r = 0; r < 8; ++r)
        dst[(size_t)r * HDIM + t * 16] = (__bf16)acc[t][r];
  } else {                                     // pure-V block; vb is [b][l][n] (transposed)
    size_t bb = (size_t)rowb >> 12;            // batch (all 8 rows share it: tiles are 128-aligned)
    size_t nn = (size_t)rowb & (NSEQ - 1);
    __bf16* dst = vb + (bb * LSEQ + (size_t)(n0 - 2 * HDIM) + col) * NSEQ + nn;
#pragma unroll
    for (int t = 0; t < 4; ++t)
#pragma unroll
      for (int r = 0; r < 8; ++r)
        dst[(size_t)t * 16 * NSEQ + r] = (__bf16)acc[t][r];
  }
}

// ---------------- Kernel 3: flash attention, 128 q-rows per workgroup, 64-key tiles ----------------
// LDS: vtT (36.9KB) + one shared buffer that is kt (17.4KB) during S and pt (18.4KB) during P@V.
// K/V tiles are staged with GLOBAL_LOAD_ASYNC_TO_LDS_B128 (no VGPR round-trip, ASYNCcnt-tracked).
__global__ __launch_bounds__(256) void flash_kernel(const __bf16* __restrict__ qb,
                                                    const __bf16* __restrict__ kb,
                                                    const __bf16* __restrict__ vb,
                                                    float* __restrict__ out) {
  __shared__ __bf16 vtT[LSEQ][PTLD];     // value tile [l][key], key-contiguous (rows 144B)
  __shared__ __bf16 sbuf[8 * 16 * PTLD]; // union: kt[64][KTLD] (S phase) / pt[8][16][PTLD] (P@V phase)

  int tid = threadIdx.x;
  int lane = tid & 31, wave = tid >> 5;
  int b = blockIdx.y;
  int row0 = blockIdx.x * 128;
  int col = lane & 15, hi = (lane >> 4) * 8;

  // q fragments for this wave's 16 rows (persist in VGPRs); Q weights carry the 1/256 scale
  const __bf16* qrow = qb + ((size_t)b * NSEQ + row0 + wave * 16) * HDIM;
  v16bf qf[4];
#pragma unroll
  for (int kk = 0; kk < 4; ++kk) qf[kk] = load_a_frag(qrow + kk * 32, HDIM, lane);

  v8f zero = {};
  v8f o[16];
#pragma unroll
  for (int t = 0; t < 16; ++t) o[t] = zero;
  float m[8], l[8];
#pragma unroll
  for (int r = 0; r < 8; ++r) { m[r] = -3.0e38f; l[r] = 0.f; }

  const __bf16* kbase = kb + (size_t)b * NSEQ * HDIM;
  const __bf16* vbase = vb + (size_t)b * LSEQ * NSEQ;

  // per-thread staging coordinates (loop-invariant)
  int kkey = tid >> 2, kc0 = (tid & 3) * 32;                     // K tile: 4 threads/row, 64B each
  unsigned klds = (unsigned)(uintptr_t)&sbuf[(size_t)kkey * KTLD + kc0];
  unsigned vlds = (unsigned)(uintptr_t)&vtT[tid][0];             // V tile: 1 thread/row, 128B each

  for (int k0 = 0; k0 < NSEQ; k0 += KT) {
    // async stage of K tile (64 keys x 128 h, h-contiguous)
    {
      const __bf16* src = kbase + (size_t)(k0 + kkey) * HDIM + kc0;
#pragma unroll
      for (int i = 0; i < 4; ++i) async_copy_b128(klds + 16u * i, src + 8 * i);
    }
    // async stage of V tile (256 l-rows x 64 keys, key-contiguous)
    {
      const __bf16* src = vbase + (size_t)tid * NSEQ + k0;
#pragma unroll
      for (int i = 0; i < 8; ++i) async_copy_b128(vlds + 16u * i, src + 8 * i);
    }
    // prefetch next tile while we compute on this one (global_prefetch_b8)
    if (k0 + KT < NSEQ) {
      __builtin_prefetch(kbase + (size_t)(k0 + KT + kkey) * HDIM + kc0, 0, 3);
      __builtin_prefetch(vbase + (size_t)tid * NSEQ + k0 + KT, 0, 3);
    }
    wait_asynccnt0();   // our tile chunks have landed in LDS
    __syncthreads();    // everyone's chunks visible to all waves

    // S = (q/256) . k^T  -> four 16x16 tiles, K-dim = 128 (4 WMMAs each)
    v8f s[4];
#pragma unroll
    for (int t = 0; t < 4; ++t) {
      v8f a = zero;
#pragma unroll
      for (int kk = 0; kk < 4; ++kk) {
        v16bf bf = load_b_frag_T(&sbuf[(size_t)(t * 16) * KTLD + kk * 32], KTLD, lane);
        a = wmma_bf16(qf[kk], bf, a);
      }
      s[t] = a;
    }

    // online softmax: rows live in 16-lane groups, one row per acc element (row = hi + r)
    float alpha[8], rs[8];
#pragma unroll
    for (int r = 0; r < 8; ++r) {
      float v = fmaxf(fmaxf(s[0][r], s[1][r]), fmaxf(s[2][r], s[3][r]));
#pragma unroll
      for (int off = 1; off < 16; off <<= 1) v = fmaxf(v, __shfl_xor(v, off, 32));
      float mn = fmaxf(m[r], v);
      alpha[r] = __expf(m[r] - mn);
      m[r] = mn;
      rs[r] = 0.f;
    }

    // all waves are done reading kt; sbuf now becomes the P staging buffer
    __syncthreads();

    __bf16* pt = &sbuf[(size_t)(wave * 16) * PTLD];
#pragma unroll
    for (int t = 0; t < 4; ++t) {
#pragma unroll
      for (int r = 0; r < 8; ++r) {
        float p = __expf(s[t][r] - m[r]);
        pt[(size_t)(hi + r) * PTLD + t * 16 + col] = (__bf16)p;
        rs[r] += p;
      }
    }
#pragma unroll
    for (int r = 0; r < 8; ++r) {
      float v = rs[r];
#pragma unroll
      for (int off = 1; off < 16; off <<= 1) v += __shfl_xor(v, off, 32);
      l[r] = l[r] * alpha[r] + v;
    }
#pragma unroll
    for (int t = 0; t < 16; ++t) {
#pragma unroll
      for (int r = 0; r < 8; ++r) o[t][r] *= alpha[r];
    }

    // drain our P stores before re-reading them as the A operand (same-wave DS is in-order)
    asm volatile("s_wait_dscnt 0" ::: "memory");

    // O += P @ V   (K-dim = 64 -> two A frags; 16 output tiles covering L = 256)
    v16bf pa0 = load_a_frag(pt, PTLD, lane);       // keys 0..31
    v16bf pa1 = load_a_frag(pt + 32, PTLD, lane);  // keys 32..63
#pragma unroll
    for (int t = 0; t < 16; ++t) {
      o[t] = wmma_bf16(pa0, load_b_frag_T(&vtT[t * 16][0], PTLD, lane), o[t]);
      o[t] = wmma_bf16(pa1, load_b_frag_T(&vtT[t * 16][32], PTLD, lane), o[t]);
    }
    __syncthreads();
  }

  // normalize and write out f32 [B,N,L]; one base pointer, all-immediate offsets
  float inv[8];
#pragma unroll
  for (int r = 0; r < 8; ++r) inv[r] = 1.0f / l[r];
  float* obase = out + ((size_t)b * NSEQ + row0 + wave * 16 + hi) * LSEQ + col;
#pragma unroll
  for (int t = 0; t < 16; ++t) {
#pragma unroll
    for (int r = 0; r < 8; ++r) {
      obase[(size_t)r * LSEQ + t * 16] = o[t][r] * inv[r];
    }
  }
}

// ---------------- host-side launch ----------------
extern "C" void kernel_launch(void* const* d_in, const int* in_sizes, int n_in,
                              void* d_out, int out_size, void* d_ws, size_t ws_size,
                              hipStream_t stream) {
  (void)in_sizes; (void)n_in; (void)out_size; (void)ws_size;
  const float* x  = (const float*)d_in[0];
  const float* Q  = (const float*)d_in[1];
  const float* K  = (const float*)d_in[2];
  const float* Vd = (const float*)d_in[3];
  const float* Vu = (const float*)d_in[4];
  float* out = (float*)d_out;

  // workspace layout (bf16): q[BN x H] | k[BN x H] | vT[B][L][N] | Wt[512 x 256]  ~= 32.25 MB
  __bf16* qb = (__bf16*)d_ws;
  __bf16* kb = qb + (size_t)BATCH * NSEQ * HDIM;
  __bf16* vb = kb + (size_t)BATCH * NSEQ * HDIM;
  __bf16* Wt = vb + (size_t)BATCH * NSEQ * LSEQ;

  prep_kernel<<<dim3((LSEQ * WCOLS + 255) / 256), 256, 0, stream>>>(Q, K, Vd, Vu, Wt);
  proj_kernel<<<dim3((BATCH * NSEQ) / 128, WCOLS / 64), 256, 0, stream>>>(x, Wt, qb, kb, vb);
  flash_kernel<<<dim3(NSEQ / 128, BATCH), 256, 0, stream>>>(qb, kb, vb, out);
}
